// TransformerBlockQuantum_65481071406795
// MI455X (gfx1250) — compile-verified, matrix operand order checked by reading, not run on hardware
//
#include <hip/hip_runtime.h>
#include <stdint.h>

typedef __attribute__((ext_vector_type(16))) _Float16 v16h;
typedef __attribute__((ext_vector_type(8)))  float    v8f;

#define EMBED 512
#define HEADS 8
#define DKQ   64
#define FFNH  2048
#define BATCH 2
#define SEQ   2048
#define BS    (BATCH * SEQ)

static __device__ __forceinline__ v8f wmma16x16x32(v16h a, v16h b, v8f c) {
  // D = A(16x32 f16) * B(32x16 f16) + C(16x16 f32)
  return __builtin_amdgcn_wmma_f32_16x16x32_f16(false, a, false, b, (short)0, c,
                                                false, false);
}

// Native base-2 exponential (v_exp_f32)
static __device__ __forceinline__ float fast_exp2(float x) {
  return __builtin_amdgcn_exp2f(x);
}

// CDNA5 async global->LDS copy (ASYNCcnt-tracked), 16 bytes per lane.
static __device__ __forceinline__ void async_copy_b128(void* lds_dst,
                                                       const void* gsrc) {
  uint32_t lo = (uint32_t)(uintptr_t)lds_dst;
  uint64_t ga = (uint64_t)(uintptr_t)gsrc;
  asm volatile("global_load_async_to_lds_b128 %0, %1, off"
               :: "v"(lo), "v"(ga) : "memory");
}
static __device__ __forceinline__ void async_wait0() {
  asm volatile("s_wait_asynccnt 0x0" ::: "memory");
}

// ---------------------------------------------------------------------------
// f32 -> f16 conversion (weights)
// ---------------------------------------------------------------------------
__global__ __launch_bounds__(256) void f32_to_f16_kernel(const float* __restrict__ s,
                                                         _Float16* __restrict__ d,
                                                         int n) {
  int i = blockIdx.x * 256 + threadIdx.x;
  if (i < n) d[i] = (_Float16)s[i];
}

// ---------------------------------------------------------------------------
// Fused LayerNorm + cos( h [+ theta[c % 64]] ) -> f16
// One wave (32 lanes) per 512-element row; wave32 shuffle reduction.
// ---------------------------------------------------------------------------
__global__ __launch_bounds__(256) void ln_cos_kernel(const float* __restrict__ x,
                                                     const float* __restrict__ g,
                                                     const float* __restrict__ be,
                                                     const float* __restrict__ theta,
                                                     _Float16* __restrict__ out) {
  int wave = threadIdx.x >> 5;
  int lane = threadIdx.x & 31;
  size_t r = (size_t)blockIdx.x * 8 + wave;
  const float* xr = x + r * EMBED;

  float vals[16];
  float s = 0.f, ss = 0.f;
#pragma unroll
  for (int i = 0; i < 16; ++i) {
    float v = xr[lane * 16 + i];
    vals[i] = v;
    s += v;
    ss += v * v;
  }
#pragma unroll
  for (int m = 1; m < 32; m <<= 1) {
    s += __shfl_xor(s, m, 32);
    ss += __shfl_xor(ss, m, 32);
  }
  const float inv = 1.0f / (float)EMBED;
  float mu = s * inv;
  float var = ss * inv - mu * mu;
  float rstd = rsqrtf(var + 1e-5f);

  _Float16* orow = out + r * EMBED;
#pragma unroll
  for (int i = 0; i < 16; ++i) {
    int c = lane * 16 + i;
    float h = (vals[i] - mu) * rstd * g[c] + be[c];
    if (theta) h += theta[c & (DKQ - 1)];
    orow[c] = (_Float16)__cosf(h);
  }
}

// ---------------------------------------------------------------------------
// Flash attention with CONSTANT-SHIFT softmax. Q = K = V = q in [-1,1]^64,
// so |score| = |q.k|/8 <= 8; softmax is shift-invariant, so use the fixed
// shift 8 instead of a running max: no max reductions, no rescaling.
// Q is pre-scaled by log2(e)/8 so p = exp2(s - 8*log2e) is one v_exp_f32.
// grid(S/64, B*H), 128 threads = 4 waves; 64-key tiles; K tile filled by
// async global->LDS; transposed copy vt gives contiguous V B-fragments.
// ---------------------------------------------------------------------------
__global__ __launch_bounds__(128) void attn_kernel(const _Float16* __restrict__ q,
                                                   _Float16* __restrict__ o) {
  const int h = blockIdx.y % HEADS;
  const int b = blockIdx.y / HEADS;
  const int wave = threadIdx.x >> 5;
  const int lane = threadIdx.x & 31;
  const int row = lane & 15;               // A-frag row
  const int col = lane & 15;               // B/C-frag column
  const int kb = (lane < 16) ? 0 : 8;      // K sub-offset within frag
  const int rhalf = (lane < 16) ? 0 : 8;   // C-frag row half

  const _Float16* qh = q + (size_t)b * SEQ * EMBED + h * DKQ;  // row stride EMBED
  const int m0 = blockIdx.x * 64 + wave * 16;

  // Q tile (16 x 64) as two A fragments, pre-scaled by log2(e)/sqrt(dk)
  v16h aq0, aq1;
  {
    const float qsc = 0.18033688011112042f;  // log2(e) / 8
    const _Float16* ab = qh + (size_t)(m0 + row) * EMBED + kb;
#pragma unroll
    for (int e = 0; e < 8; ++e) {
      aq0[e] = (_Float16)((float)ab[e] * qsc);
      aq0[e + 8] = (_Float16)((float)ab[16 + e] * qsc);
      aq1[e] = (_Float16)((float)ab[32 + e] * qsc);
      aq1[e + 8] = (_Float16)((float)ab[48 + e] * qsc);
    }
  }

  __shared__ _Float16 kt[64][DKQ + 8];   // K tile, row-major (key, dk)
  __shared__ _Float16 vt[DKQ][64 + 8];   // V tile, transposed (dk, key)
  __shared__ _Float16 pt[4][16][64 + 8]; // per-wave P staging (16x64)

  v8f acc[4] = {};                        // 16 x 64 output accumulator
  float rl[8];
#pragma unroll
  for (int j = 0; j < 8; ++j) rl[j] = 0.f;

  const int tr = threadIdx.x >> 1;        // 0..63 : key row
  const int tc = (threadIdx.x & 1) * 32;  // 0/32  : dk column base

  for (int k0 = 0; k0 < SEQ; k0 += 64) {
    __syncthreads();  // previous tile fully consumed
    {
      const _Float16* src = qh + (size_t)(k0 + tr) * EMBED + tc;
      // async fill of the row-major K tile (4 x 16B per thread)
      async_copy_b128(&kt[tr][tc], src);
      async_copy_b128(&kt[tr][tc + 8], src + 8);
      async_copy_b128(&kt[tr][tc + 16], src + 16);
      async_copy_b128(&kt[tr][tc + 24], src + 24);
      // register load + scatter for the transposed V tile
      _Float16 vreg[32];
#pragma unroll
      for (int e = 0; e < 32; ++e) vreg[e] = src[e];
#pragma unroll
      for (int e = 0; e < 32; ++e) vt[tc + e][tr] = vreg[e];
    }
    async_wait0();
    __syncthreads();  // tile ready

    // -------- scores: S(16 x 64) = Q(16x64) * K^T(64x64), four C tiles -----
    v8f c[4] = {};
#pragma unroll
    for (int t = 0; t < 4; ++t) {
      v16h bk;
#pragma unroll
      for (int e = 0; e < 8; ++e) {
        bk[e] = kt[t * 16 + col][kb + e];
        bk[e + 8] = kt[t * 16 + col][16 + kb + e];
      }
      c[t] = wmma16x16x32(aq0, bk, c[t]);
#pragma unroll
      for (int e = 0; e < 8; ++e) {
        bk[e] = kt[t * 16 + col][32 + kb + e];
        bk[e + 8] = kt[t * 16 + col][48 + kb + e];
      }
      c[t] = wmma16x16x32(aq1, bk, c[t]);
    }

    // -------- constant-shift softmax numerators: p = exp2(s - 8*log2e) -----
    const float M2 = 11.541560327111708f;  // 8 * log2(e)
#pragma unroll
    for (int j = 0; j < 8; ++j) {
      float p0 = fast_exp2(c[0][j] - M2);
      float p1 = fast_exp2(c[1][j] - M2);
      float p2 = fast_exp2(c[2][j] - M2);
      float p3 = fast_exp2(c[3][j] - M2);
      rl[j] += (p0 + p1) + (p2 + p3);      // per-lane partial row sum
      pt[wave][j + rhalf][col] = (_Float16)p0;
      pt[wave][j + rhalf][16 + col] = (_Float16)p1;
      pt[wave][j + rhalf][32 + col] = (_Float16)p2;
      pt[wave][j + rhalf][48 + col] = (_Float16)p3;
    }

    // re-load P as two A fragments (C layout -> A layout via LDS)
    v16h ap0, ap1;
#pragma unroll
    for (int e = 0; e < 8; ++e) {
      ap0[e] = pt[wave][row][kb + e];
      ap0[e + 8] = pt[wave][row][16 + kb + e];
      ap1[e] = pt[wave][row][32 + kb + e];
      ap1[e + 8] = pt[wave][row][48 + kb + e];
    }

    // -------- P(16x64) * V(64x64): contiguous b128 loads from vt -----------
#pragma unroll
    for (int n = 0; n < 4; ++n) {
      v16h bv;
#pragma unroll
      for (int e = 0; e < 8; ++e) {
        bv[e] = vt[n * 16 + col][kb + e];
        bv[e + 8] = vt[n * 16 + col][16 + kb + e];
      }
      acc[n] = wmma16x16x32(ap0, bv, acc[n]);
#pragma unroll
      for (int e = 0; e < 8; ++e) {
        bv[e] = vt[n * 16 + col][32 + kb + e];
        bv[e + 8] = vt[n * 16 + col][48 + kb + e];
      }
      acc[n] = wmma16x16x32(ap1, bv, acc[n]);
    }
  }

  // -------- one deferred row-sum reduction, then normalize + store ---------
#pragma unroll
  for (int j = 0; j < 8; ++j) {
    float s = rl[j];
#pragma unroll
    for (int m = 1; m < 16; m <<= 1) s += __shfl_xor(s, m, 32);
    rl[j] = 1.0f / s;
  }
  _Float16* ob = o + (size_t)b * SEQ * EMBED + h * DKQ;
#pragma unroll
  for (int n = 0; n < 4; ++n) {
#pragma unroll
    for (int j = 0; j < 8; ++j) {
      int r = m0 + j + rhalf;
      ob[(size_t)r * EMBED + n * 16 + col] = (_Float16)(acc[n][j] * rl[j]);
    }
  }
}

// ---------------------------------------------------------------------------
// WMMA GEMM: C[m][n] = act( sum_k A[m][k] * W[n][k] + bias[n] ) (+ resid)
// A: (BS x K) f16 row-major, W: (N x K) f16 row-major.
// 256 threads = 8 waves; block tile 256(M) x 64(N); wave tile 32 x 64:
// 2 A-frags + 4 B-frags feed 8 WMMAs per 32-wide K step.
// ---------------------------------------------------------------------------
template <bool RELU, bool RES, typename OutT>
__global__ __launch_bounds__(256) void gemm_kernel(const _Float16* __restrict__ A,
                                                   const _Float16* __restrict__ W,
                                                   const float* __restrict__ bias,
                                                   const float* __restrict__ resid,
                                                   OutT* __restrict__ C,
                                                   int N, int K) {
  const int wave = threadIdx.x >> 5;
  const int lane = threadIdx.x & 31;
  const int row = lane & 15;
  const int col = lane & 15;
  const int kb = (lane < 16) ? 0 : 8;
  const int rhalf = (lane < 16) ? 0 : 8;

  const int m0 = (blockIdx.y * 8 + wave) * 32;
  const int n0 = blockIdx.x * 64;

  v8f acc[2][4] = {};

  for (int k0 = 0; k0 < K; k0 += 32) {
    v16h a0, a1;
    const _Float16* ab0 = A + (size_t)(m0 + row) * K + k0 + kb;
    const _Float16* ab1 = ab0 + (size_t)16 * K;
#pragma unroll
    for (int e = 0; e < 8; ++e) {
      a0[e] = ab0[e]; a0[e + 8] = ab0[16 + e];
      a1[e] = ab1[e]; a1[e + 8] = ab1[16 + e];
    }
    __builtin_prefetch(ab0 + 32, 0, 1);  // next K tile of A
    __builtin_prefetch(ab1 + 32, 0, 1);

#pragma unroll
    for (int n = 0; n < 4; ++n) {
      v16h bf;
      const _Float16* wb = W + (size_t)(n0 + n * 16 + col) * K + k0 + kb;
#pragma unroll
      for (int e = 0; e < 8; ++e) { bf[e] = wb[e]; bf[e + 8] = wb[16 + e]; }
      acc[0][n] = wmma16x16x32(a0, bf, acc[0][n]);
      acc[1][n] = wmma16x16x32(a1, bf, acc[1][n]);
    }
  }

#pragma unroll
  for (int mi = 0; mi < 2; ++mi) {
#pragma unroll
    for (int n = 0; n < 4; ++n) {
      int c = n0 + n * 16 + col;
      float bv = bias[c];
#pragma unroll
      for (int j = 0; j < 8; ++j) {
        size_t idx = (size_t)(m0 + mi * 16 + j + rhalf) * N + c;
        float v = acc[mi][n][j] + bv;
        if (RELU) v = fmaxf(v, 0.f);
        if (RES) v += resid[idx];
        C[idx] = (OutT)v;
      }
    }
  }
}

// ---------------------------------------------------------------------------
// Host launch
// ---------------------------------------------------------------------------
extern "C" void kernel_launch(void* const* d_in, const int* in_sizes, int n_in,
                              void* d_out, int out_size, void* d_ws, size_t ws_size,
                              hipStream_t stream) {
  (void)in_sizes; (void)n_in; (void)out_size; (void)ws_size;
  const float* x    = (const float*)d_in[0];
  const float* g1   = (const float*)d_in[1];
  const float* be1  = (const float*)d_in[2];
  const float* th_a = (const float*)d_in[3];
  const float* Wo   = (const float*)d_in[4];
  const float* bo   = (const float*)d_in[5];
  const float* g2   = (const float*)d_in[6];
  const float* be2  = (const float*)d_in[7];
  // d_in[8] = theta_ffn: RZ commutes with Z measurement, no effect on output
  const float* W1   = (const float*)d_in[9];
  const float* bf1  = (const float*)d_in[10];
  const float* W2   = (const float*)d_in[11];
  const float* bf2  = (const float*)d_in[12];

  // workspace carve-up
  _Float16* qh  = (_Float16*)d_ws;                       // (BS x D) f16
  _Float16* ao  = qh + (size_t)BS * EMBED;               // attn out f16
  float*    x1  = (float*)(ao + (size_t)BS * EMBED);     // residual 1, f32
  _Float16* qo  = (_Float16*)(x1 + (size_t)BS * EMBED);  // cos(ln2) f16
  _Float16* hid = qo + (size_t)BS * EMBED;               // (BS x FFN) f16
  _Float16* WoH = hid + (size_t)BS * FFNH;
  _Float16* W1H = WoH + (size_t)EMBED * EMBED;
  _Float16* W2H = W1H + (size_t)FFNH * EMBED;

  f32_to_f16_kernel<<<(EMBED * EMBED + 255) / 256, 256, 0, stream>>>(Wo, WoH, EMBED * EMBED);
  f32_to_f16_kernel<<<(FFNH * EMBED + 255) / 256, 256, 0, stream>>>(W1, W1H, FFNH * EMBED);
  f32_to_f16_kernel<<<(EMBED * FFNH + 255) / 256, 256, 0, stream>>>(W2, W2H, EMBED * FFNH);

  // q = cos(LN(x) + theta_attn)
  ln_cos_kernel<<<BS / 8, 256, 0, stream>>>(x, g1, be1, th_a, qh);

  // self-attention (Q=K=V=q)
  attn_kernel<<<dim3(SEQ / 64, BATCH * HEADS), 128, 0, stream>>>(qh, ao);

  // x1 = x + ao @ Wo^T + bo
  gemm_kernel<false, true, float>
      <<<dim3(EMBED / 64, BS / 256), 256, 0, stream>>>(ao, WoH, bo, x, x1, EMBED, EMBED);

  // qout = cos(LN(x1))
  ln_cos_kernel<<<BS / 8, 256, 0, stream>>>(x1, g2, be2, nullptr, qo);

  // hid = relu(qout @ W1^T + bf1)
  gemm_kernel<true, false, _Float16>
      <<<dim3(FFNH / 64, BS / 256), 256, 0, stream>>>(qo, W1H, bf1, nullptr, hid, FFNH, EMBED);

  // out = x1 + hid @ W2^T + bf2
  gemm_kernel<false, true, float>
      <<<dim3(EMBED / 64, BS / 256), 256, 0, stream>>>(hid, W2H, bf2, x1, (float*)d_out, EMBED, FFNH);
}
